// VQVAE_85212151152778
// MI455X (gfx1250) — compile-verified, matrix operand order checked by reading, not run on hardware
//
#include <hip/hip_runtime.h>
#include <hip/hip_bf16.h>

// ---------------------------------------------------------------------------
// VQ-VAE forward for MI455X (gfx1250): bf16 WMMA GEMMs with f32 accumulation,
// double-buffered LDS tiles staged with async global->LDS copies (ASYNCcnt).
// Compute-bound (~470 GFLOP vs ~600MB HBM traffic).
// ---------------------------------------------------------------------------

#define SEGN 65536
#define LATD 128
#define KCODES 1024

typedef __attribute__((ext_vector_type(16))) __bf16 v16bf;
typedef __attribute__((ext_vector_type(8)))  float  v8f;

union Frag16 { v16bf v; unsigned int u[8]; };

// ---- async global->LDS support probe (gfx1250) ----
#if defined(__has_builtin)
#if __has_builtin(__builtin_amdgcn_global_load_async_to_lds_b128)
#define VQ_HAVE_ASYNC 1
#endif
#endif

#if VQ_HAVE_ASYNC
// Builtin prototype (from clang diagnostic): first arg is
// 'int __vector_size__(4*sizeof(int)) __device__ *'  -> v4i in addrspace(1).
typedef int v4i __attribute__((vector_size(4 * sizeof(int))));
typedef __attribute__((address_space(1))) v4i glb_v4i;
typedef __attribute__((address_space(3))) v4i lds_v4i;
#if __has_builtin(__builtin_amdgcn_s_wait_asynccnt)
#define VQ_WAIT_ASYNC() __builtin_amdgcn_s_wait_asynccnt(0)
#else
#define VQ_WAIT_ASYNC() asm volatile("s_wait_asynccnt 0x0" ::: "memory")
#endif
#endif

// LDS tile row pitch: 40 bf16 = 80 bytes.  Rows are 16B-aligned (b128 async
// stores legal) and 20-dword stride -> 20*l16 mod 64 distinct for l16=0..15,
// so fragment reads are bank-conflict-free.
#define APITCH 40

// ---------------------------------------------------------------------------
// Conversion helpers
// ---------------------------------------------------------------------------
__global__ void vq_convert_transpose(const float* __restrict__ W,
                                     __bf16* __restrict__ Wt, int I, int O) {
  int t = blockIdx.x * 256 + threadIdx.x;
  if (t < I * O) {
    int i = t / O, o = t - i * O;
    Wt[(size_t)o * I + i] = (__bf16)W[t];
  }
}

__global__ void vq_convert(const float* __restrict__ src,
                           __bf16* __restrict__ dst, int n) {
  int t = blockIdx.x * 256 + threadIdx.x;
  if (t < n) dst[t] = (__bf16)src[t];
}

__global__ void vq_enorm(const float* __restrict__ E, float* __restrict__ en) {
  int c = blockIdx.x * 256 + threadIdx.x;
  if (c < KCODES) {
    float s = 0.f;
    const float* row = E + (size_t)c * LATD;
    for (int d = 0; d < LATD; ++d) { float v = row[d]; s += v * v; }
    en[c] = s;
  }
}

// ---------------------------------------------------------------------------
// Tiled GEMM:  C[M,O] = act(A[M,K] @ W[K,O] + b)
// Weights pre-transposed to Wt[O][K] bf16.  Block tile 128x64, K-step 32,
// double-buffered LDS with async global->LDS staging.  8 waves (4 M x 2 N),
// each wave 32x32 via 2x2 v_wmma_f32_16x16x32_bf16.
// ---------------------------------------------------------------------------
template<bool A_IS_F32, bool RELU, bool WRITE_F32, bool WRITE_BF16>
__global__ __launch_bounds__(256)
void vq_gemm_wmma(const void* __restrict__ Aptr,
                  const __bf16* __restrict__ Wt,
                  const float* __restrict__ bias,
                  float* __restrict__ Cf,
                  __bf16* __restrict__ Cb,
                  int Kdim, int Odim) {
  __shared__ __bf16 lsA[2][128 * APITCH];
  __shared__ __bf16 lsB[2][64 * APITCH];

  const float*  Af = (const float*)Aptr;
  const __bf16* Ab = (const __bf16*)Aptr;

  const int t     = threadIdx.x;
  const int wid   = t >> 5;
  const int lane  = t & 31;
  const int half  = lane >> 4;
  const int l16   = lane & 15;
  const int waveM = wid & 3;   // 0..3
  const int waveN = wid >> 2;  // 0..1

  const int rowBase = blockIdx.y * 128;
  const int colBase = blockIdx.x * 64;

  // staging assignments
  const int ar  = t >> 1;          // A tile row 0..127
  const int acs = (t & 1) * 16;    // A col segment (16 elems = 32B)
  const int br  = t >> 2;          // B tile row 0..63 (output dim)
  const int bcs = (t & 3) * 8;     // B col segment (8 elems = 16B)

  auto stageA = [&](int buf, int kt) {
    if (A_IS_F32) {
      // synchronous convert-on-stage (async copies cannot convert f32->bf16)
      const float4* src =
          (const float4*)(Af + (size_t)(rowBase + ar) * Kdim + kt + acs);
      __bf16* bd = &lsA[buf][ar * APITCH + acs];
      #pragma unroll
      for (int j = 0; j < 4; ++j) {
        float4 v = src[j];
        bd[j*4+0] = (__bf16)v.x; bd[j*4+1] = (__bf16)v.y;
        bd[j*4+2] = (__bf16)v.z; bd[j*4+3] = (__bf16)v.w;
      }
    } else {
      const __bf16* g = Ab + (size_t)(rowBase + ar) * Kdim + kt + acs;
      __bf16* l = &lsA[buf][ar * APITCH + acs];
#if VQ_HAVE_ASYNC
      __builtin_amdgcn_global_load_async_to_lds_b128(
          (glb_v4i*)g, (lds_v4i*)l, 0, 0);
      __builtin_amdgcn_global_load_async_to_lds_b128(
          (glb_v4i*)g, (lds_v4i*)l, 16, 0);
#else
      const uint4* s128 = (const uint4*)g;
      unsigned int* d32 = (unsigned int*)l;
      uint4 u0 = s128[0], u1 = s128[1];
      d32[0] = u0.x; d32[1] = u0.y; d32[2] = u0.z; d32[3] = u0.w;
      d32[4] = u1.x; d32[5] = u1.y; d32[6] = u1.z; d32[7] = u1.w;
#endif
    }
  };

  auto stageB = [&](int buf, int kt) {
    const __bf16* g = Wt + (size_t)(colBase + br) * Kdim + kt + bcs;
    __bf16* l = &lsB[buf][br * APITCH + bcs];
#if VQ_HAVE_ASYNC
    __builtin_amdgcn_global_load_async_to_lds_b128(
        (glb_v4i*)g, (lds_v4i*)l, 0, 0);
#else
    const uint4* s128 = (const uint4*)g;
    unsigned int* d32 = (unsigned int*)l;
    uint4 u = *s128;
    d32[0] = u.x; d32[1] = u.y; d32[2] = u.z; d32[3] = u.w;
#endif
  };

  v8f acc[2][2] = {};

  // prologue: stage first K tile into buffer 0
  stageA(0, 0);
  stageB(0, 0);

  for (int kt = 0, it = 0; kt < Kdim; kt += 32, ++it) {
    const int buf = it & 1;
#if VQ_HAVE_ASYNC
    VQ_WAIT_ASYNC();           // my async stores into lsX[buf] are complete
#endif
    __syncthreads();           // everyone's staging of lsX[buf] visible

    // kick off the next tile while we compute on this one
    if (kt + 32 < Kdim) {
      stageA(buf ^ 1, kt + 32);
      stageB(buf ^ 1, kt + 32);
    }

    // ---- fragments ----
    Frag16 afr[2], bfr[2];
    #pragma unroll
    for (int mi = 0; mi < 2; ++mi) {
      const __bf16* base = &lsA[buf][(waveM * 32 + mi * 16 + l16) * APITCH];
      #pragma unroll
      for (int v = 0; v < 8; ++v) {
        int k = ((v < 4) ? 0 : 16) + half * 8 + (v & 3) * 2;
        afr[mi].u[v] = *(const unsigned int*)&base[k];
      }
    }
    #pragma unroll
    for (int ni = 0; ni < 2; ++ni) {
      const __bf16* base = &lsB[buf][(waveN * 32 + ni * 16 + l16) * APITCH];
      #pragma unroll
      for (int v = 0; v < 8; ++v) {
        int k = ((v < 4) ? 0 : 16) + half * 8 + (v & 3) * 2;
        bfr[ni].u[v] = *(const unsigned int*)&base[k];
      }
    }
    #pragma unroll
    for (int mi = 0; mi < 2; ++mi)
      #pragma unroll
      for (int ni = 0; ni < 2; ++ni)
        acc[mi][ni] = __builtin_amdgcn_wmma_f32_16x16x32_bf16(
            false, afr[mi].v, false, bfr[ni].v,
            (short)0, acc[mi][ni], false, false);

    __syncthreads();           // all waves done reading lsX[buf]
  }

  // ---- epilogue: bias + relu + store (f32 and/or bf16) ----
  #pragma unroll
  for (int mi = 0; mi < 2; ++mi) {
    #pragma unroll
    for (int ni = 0; ni < 2; ++ni) {
      int m0  = rowBase + waveM * 32 + mi * 16 + 8 * half;
      int col = colBase + waveN * 32 + ni * 16 + l16;
      float bc = bias[col];
      #pragma unroll
      for (int r = 0; r < 8; ++r) {
        float v = acc[mi][ni][r] + bc;
        if (RELU) v = fmaxf(v, 0.f);
        size_t o = (size_t)(m0 + r) * Odim + col;
        if (WRITE_F32)  Cf[o] = v;
        if (WRITE_BF16) Cb[o] = (__bf16)v;
      }
    }
  }
}

// ---------------------------------------------------------------------------
// VQ argmin: block = 16 samples x all 1024 codes.
// score(n,c) = ||e_c||^2 - 2 * z_n . e_c   (||z||^2 is row-constant).
// Each wave owns 128 codes: 8 fragments x 4 K-steps of WMMA; argmin via
// per-lane running min -> shfl_xor cross-lane reduce -> small LDS combine.
// ---------------------------------------------------------------------------
__global__ __launch_bounds__(256)
void vq_argmin_wmma(const __bf16* __restrict__ zb,
                    const __bf16* __restrict__ Eb,
                    const float* __restrict__ en,
                    int* __restrict__ idxOut) {
  __shared__ __bf16 zt[16 * 132];  // 66-dword pitch -> conflict-free
  __shared__ float redV[8 * 16];
  __shared__ int   redI[8 * 16];

  const int t  = threadIdx.x;
  const int s0 = blockIdx.x * 16;

  // stage z tile 16 x 128 bf16
  {
    int sample = t >> 4;
    int seg    = (t & 15) * 8;
    uint4 d = *(const uint4*)(zb + (size_t)(s0 + sample) * LATD + seg);
    unsigned int* dst = (unsigned int*)&zt[sample * 132 + seg];
    dst[0] = d.x; dst[1] = d.y; dst[2] = d.z; dst[3] = d.w;
  }
  __syncthreads();

  const int wid  = t >> 5;
  const int lane = t & 31;
  const int half = lane >> 4;
  const int l16  = lane & 15;

  v8f acc[8] = {};
  const unsigned int* Eu = (const unsigned int*)Eb;

  for (int ks = 0; ks < 4; ++ks) {
    Frag16 a;
    #pragma unroll
    for (int v = 0; v < 8; ++v) {
      int k = ks * 32 + ((v < 4) ? 0 : 16) + half * 8 + (v & 3) * 2;
      a.u[v] = *(const unsigned int*)&zt[l16 * 132 + k];
    }
    #pragma unroll
    for (int ni = 0; ni < 8; ++ni) {
      int code = wid * 128 + ni * 16 + l16;
      Frag16 b;
      const unsigned int* ep = Eu + (size_t)code * (LATD / 2);
      #pragma unroll
      for (int v = 0; v < 8; ++v) {
        int k = ks * 32 + ((v < 4) ? 0 : 16) + half * 8 + (v & 3) * 2;
        b.u[v] = ep[k >> 1];
      }
      acc[ni] = __builtin_amdgcn_wmma_f32_16x16x32_bf16(
          false, a.v, false, b.v, (short)0, acc[ni], false, false);
    }
  }

  // per-lane running argmin over this wave's 128 codes
  float bv[8]; int bi[8];
  #pragma unroll
  for (int r = 0; r < 8; ++r) { bv[r] = 3.402823466e38f; bi[r] = 0x7fffffff; }
  #pragma unroll
  for (int ni = 0; ni < 8; ++ni) {
    int code = wid * 128 + ni * 16 + l16;
    float e2 = en[code];
    #pragma unroll
    for (int r = 0; r < 8; ++r) {
      float sc = e2 - 2.0f * acc[ni][r];
      if (sc < bv[r] || (sc == bv[r] && code < bi[r])) { bv[r] = sc; bi[r] = code; }
    }
  }
  // cross-lane min within each 16-lane half (codes vary across l16)
  #pragma unroll
  for (int m = 1; m < 16; m <<= 1) {
    #pragma unroll
    for (int r = 0; r < 8; ++r) {
      float ov = __shfl_xor(bv[r], m, 32);
      int   oi = __shfl_xor(bi[r], m, 32);
      if (ov < bv[r] || (ov == bv[r] && oi < bi[r])) { bv[r] = ov; bi[r] = oi; }
    }
  }
  if (l16 == 0) {
    #pragma unroll
    for (int r = 0; r < 8; ++r) {
      int sample = r + 8 * half;
      redV[wid * 16 + sample] = bv[r];
      redI[wid * 16 + sample] = bi[r];
    }
  }
  __syncthreads();
  if (t < 16) {
    float best = redV[t]; int bid = redI[t];
    for (int w = 1; w < 8; ++w) {
      float v = redV[w * 16 + t]; int i = redI[w * 16 + t];
      if (v < best || (v == best && i < bid)) { best = v; bid = i; }
    }
    idxOut[s0 + t] = bid;
  }
}

// ---------------------------------------------------------------------------
// Gather z_q = E[idx]; write f32 to output and bf16 for the decoder.
// ---------------------------------------------------------------------------
__global__ void vq_gather(const int* __restrict__ idx,
                          const float* __restrict__ E,
                          float* __restrict__ zq_f,
                          __bf16* __restrict__ zq_b) {
  int t = blockIdx.x * 256 + threadIdx.x;   // N*128 threads
  int n = t >> 7, d = t & 127;
  int id = idx[n];
  float e = E[(size_t)id * LATD + d];
  zq_f[t] = e;
  zq_b[t] = (__bf16)e;
}

// ---------------------------------------------------------------------------
// Launcher
// ---------------------------------------------------------------------------
extern "C" void kernel_launch(void* const* d_in, const int* in_sizes, int n_in,
                              void* d_out, int out_size, void* d_ws, size_t ws_size,
                              hipStream_t stream) {
  const int N = SEGN;
  const float* x   = (const float*)d_in[0];
  const float* We0 = (const float*)d_in[1];  const float* be0 = (const float*)d_in[2];
  const float* We1 = (const float*)d_in[3];  const float* be1 = (const float*)d_in[4];
  const float* We2 = (const float*)d_in[5];  const float* be2 = (const float*)d_in[6];
  const float* We3 = (const float*)d_in[7];  const float* be3 = (const float*)d_in[8];
  const float* E   = (const float*)d_in[9];
  const float* Wd0 = (const float*)d_in[10]; const float* bd0 = (const float*)d_in[11];
  const float* Wd1 = (const float*)d_in[12]; const float* bd1 = (const float*)d_in[13];
  const float* Wd2 = (const float*)d_in[14]; const float* bd2 = (const float*)d_in[15];
  const float* Wd3 = (const float*)d_in[16]; const float* bd3 = (const float*)d_in[17];

  float* out  = (float*)d_out;
  float* xrec = out;                              // N x 1024
  float* zf   = out + (size_t)N * 1024;           // N x 128
  float* zqf  = zf  + (size_t)N * 128;            // N x 128

  // workspace carve-out
  char* w = (char*)d_ws;
  size_t off = 0;
  auto take = [&](size_t bytes) -> void* {
    void* p = w + off;
    off = (off + bytes + 255) & ~(size_t)255;
    return p;
  };
  __bf16* bufA = (__bf16*)take((size_t)N * 1024 * 2);  // 128 MB ping
  __bf16* bufB = (__bf16*)take((size_t)N * 1024 * 2);  // 128 MB pong
  __bf16* We0t = (__bf16*)take((size_t)1024 * 1024 * 2);
  __bf16* We1t = (__bf16*)take((size_t)512  * 1024 * 2);
  __bf16* We2t = (__bf16*)take((size_t)256  * 512  * 2);
  __bf16* We3t = (__bf16*)take((size_t)128  * 256  * 2);
  __bf16* Wd0t = (__bf16*)take((size_t)256  * 128  * 2);
  __bf16* Wd1t = (__bf16*)take((size_t)512  * 256  * 2);
  __bf16* Wd2t = (__bf16*)take((size_t)1024 * 512  * 2);
  __bf16* Wd3t = (__bf16*)take((size_t)1024 * 1024 * 2);
  __bf16* Eb   = (__bf16*)take((size_t)KCODES * LATD * 2);
  float*  enB  = (float*)take((size_t)KCODES * 4);
  int*    idx  = (int*)take((size_t)N * 4);

  dim3 blk(256);
  auto cgrid = [](int n) { return dim3((n + 255) / 256); };

  // --- weight / codebook conversion (bf16, transposed) ---
  vq_convert_transpose<<<cgrid(1024*1024), blk, 0, stream>>>(We0, We0t, 1024, 1024);
  vq_convert_transpose<<<cgrid(1024*512),  blk, 0, stream>>>(We1, We1t, 1024, 512);
  vq_convert_transpose<<<cgrid(512*256),   blk, 0, stream>>>(We2, We2t, 512, 256);
  vq_convert_transpose<<<cgrid(256*128),   blk, 0, stream>>>(We3, We3t, 256, 128);
  vq_convert_transpose<<<cgrid(128*256),   blk, 0, stream>>>(Wd0, Wd0t, 128, 256);
  vq_convert_transpose<<<cgrid(256*512),   blk, 0, stream>>>(Wd1, Wd1t, 256, 512);
  vq_convert_transpose<<<cgrid(512*1024),  blk, 0, stream>>>(Wd2, Wd2t, 512, 1024);
  vq_convert_transpose<<<cgrid(1024*1024), blk, 0, stream>>>(Wd3, Wd3t, 1024, 1024);
  vq_convert<<<cgrid(KCODES*LATD), blk, 0, stream>>>(E, Eb, KCODES * LATD);
  vq_enorm<<<cgrid(KCODES), blk, 0, stream>>>(E, enB);

  auto g = [](int M, int O) { return dim3(O / 64, M / 128); };

  // --- encoder ---
  vq_gemm_wmma<true,  true,  false, true ><<<g(N,1024), blk, 0, stream>>>(x,    We0t, be0, nullptr, bufA, 1024, 1024);
  vq_gemm_wmma<false, true,  false, true ><<<g(N, 512), blk, 0, stream>>>(bufA, We1t, be1, nullptr, bufB, 1024,  512);
  vq_gemm_wmma<false, true,  false, true ><<<g(N, 256), blk, 0, stream>>>(bufB, We2t, be2, nullptr, bufA,  512,  256);
  vq_gemm_wmma<false, false, true,  true ><<<g(N, 128), blk, 0, stream>>>(bufA, We3t, be3, zf,      bufB,  256,  128);

  // --- vector quantization ---
  vq_argmin_wmma<<<dim3(N / 16), blk, 0, stream>>>(bufB, Eb, enB, idx);
  vq_gather<<<dim3((N * 128) / 256), blk, 0, stream>>>(idx, E, zqf, bufA);

  // --- decoder ---
  vq_gemm_wmma<false, true,  false, true ><<<g(N, 256), blk, 0, stream>>>(bufA, Wd0t, bd0, nullptr, bufB,  128,  256);
  vq_gemm_wmma<false, true,  false, true ><<<g(N, 512), blk, 0, stream>>>(bufB, Wd1t, bd1, nullptr, bufA,  256,  512);
  vq_gemm_wmma<false, true,  false, true ><<<g(N,1024), blk, 0, stream>>>(bufA, Wd2t, bd2, nullptr, bufB,  512, 1024);
  vq_gemm_wmma<false, false, true,  false><<<g(N,1024), blk, 0, stream>>>(bufB, Wd3t, bd3, xrec, nullptr, 1024, 1024);

  (void)in_sizes; (void)n_in; (void)out_size; (void)ws_size;
}
